// AH_69982197121807
// MI455X (gfx1250) — matrix operand order repaired and checked by reading, not compiled, not run on hardware
//
#include <hip/hip_runtime.h>
#include <stdint.h>

// ---------------- types for WMMA / async ----------------
typedef __attribute__((ext_vector_type(16))) __bf16 v16bf;
typedef __attribute__((ext_vector_type(8)))  float  v8f;
typedef __attribute__((ext_vector_type(4)))  int    v4i;

#if defined(__AMDGCN__) && __has_builtin(__builtin_amdgcn_global_load_async_to_lds_b128)
#define USE_ASYNC_LDS 1
typedef __attribute__((address_space(1))) v4i* gv4i_p;   // global AS int4*
typedef __attribute__((address_space(3))) v4i* lv4i_p;   // LDS AS int4*
#else
#define USE_ASYNC_LDS 0
#endif

__device__ __forceinline__ uint16_t bf16_rn(float f) {
  union { float f; uint32_t u; } c; c.f = f;
  uint32_t u = c.u;
  if ((u & 0x7fffffffu) > 0x7f800000u) return (uint16_t)0x7fc0; // NaN
  u += 0x7fffu + ((u >> 16) & 1u);                              // round-nearest-even
  return (uint16_t)(u >> 16);
}

// ---------------- fp32 -> bf16 elementwise ----------------
__global__ __launch_bounds__(256) void AH_cvt_bf16(const float* __restrict__ in,
                                                   uint16_t* __restrict__ out, int n) {
  int i = blockIdx.x * 256 + threadIdx.x;
  if (i < n) out[i] = bf16_rn(in[i]);
}

// ---------------- W[K,N] fp32 -> Wt[N,K] bf16 (tiled transpose) ----------------
__global__ __launch_bounds__(256) void AH_transpose_bf16(const float* __restrict__ W,
                                                         uint16_t* __restrict__ Wt,
                                                         int K, int N) {
  __shared__ float tile[32][33];
  const int bx = blockIdx.x * 32;           // N
  const int by = blockIdx.y * 32;           // K
  const int tx = threadIdx.x & 31;
  const int ty = threadIdx.x >> 5;          // 0..7
  #pragma unroll
  for (int i = 0; i < 32; i += 8)
    tile[ty + i][tx] = W[(size_t)(by + ty + i) * N + (bx + tx)];
  __syncthreads();
  #pragma unroll
  for (int i = 0; i < 32; i += 8)
    Wt[(size_t)(bx + ty + i) * K + (by + tx)] = bf16_rn(tile[tx][ty + i]);
}

// ---------------- bf16 WMMA GEMM ----------------
// C = act(A[M,K] @ Bt[N,K]^T + bias)   (RAW=1: raw f32 partials, no bias/act)
// ACT: 0=none 1=relu 2=silu ; OUTBF16: bf16 vs f32 out
#define TM 128
#define TN 128
#define TKS 64
#define LDSROW 72   // bf16 elems/row (144B): lane-start dwords i*36 mod 64 distinct -> conflict-free

template <int ACT, int OUTBF16, int RAW>
__global__ __launch_bounds__(256) void AH_gemm_bf16(const uint16_t* __restrict__ A,
                                                    const uint16_t* __restrict__ Bt,
                                                    const float* __restrict__ bias,
                                                    void* __restrict__ Cout,
                                                    int M, int N, int K,
                                                    int kLen, size_t sliceBytes) {
  __shared__ __align__(16) uint16_t As[TM * LDSROW];
  __shared__ __align__(16) uint16_t Bs[TN * LDSROW];

  const int tid    = threadIdx.x;
  const int lane   = tid & 31;
  const int wid    = tid >> 5;
  const int waveM  = (wid & 3) * 32;        // 4 waves along M
  const int waveN  = (wid >> 2) * 64;       // 2 waves along N
  const int blockM = blockIdx.y * TM;
  const int blockN = blockIdx.x * TN;
  const int kstart = blockIdx.z * kLen;

  v8f acc[2][4];
  #pragma unroll
  for (int i = 0; i < 2; ++i)
    #pragma unroll
    for (int j = 0; j < 4; ++j)
      #pragma unroll
      for (int r = 0; r < 8; ++r) acc[i][j][r] = 0.0f;

  const int m  = lane & 15;
  const int ka = (lane >> 4) * 8;    // A frag: dw0-3 K=[ka,ka+8), dw4-7 K=[ka+16,ka+24)
  const int kb = (lane >> 4) * 8;    // B frag: dw0-3 K=[kb,kb+8), dw4-7 K=[kb+16,kb+24)

  for (int k0 = kstart; k0 < kstart + kLen; k0 += TKS) {
    // stage 128x64 A tile + 128x64 B tile into LDS: 1024 16B chunks each, 4/thread
    #pragma unroll
    for (int it = 0; it < 4; ++it) {
      const int c   = tid + it * 256;
      const int row = c >> 3;
      const int kk  = (c & 7) * 8;
#if USE_ASYNC_LDS
      __builtin_amdgcn_global_load_async_to_lds_b128(
          (gv4i_p)(uintptr_t)(A + (size_t)(blockM + row) * K + k0 + kk),
          (lv4i_p)&As[row * LDSROW + kk], 0, 0);
      __builtin_amdgcn_global_load_async_to_lds_b128(
          (gv4i_p)(uintptr_t)(Bt + (size_t)(blockN + row) * K + k0 + kk),
          (lv4i_p)&Bs[row * LDSROW + kk], 0, 0);
#else
      uint4 va = *(const uint4*)(A  + (size_t)(blockM + row) * K + k0 + kk);
      *(uint4*)(&As[row * LDSROW + kk]) = va;
      uint4 vb = *(const uint4*)(Bt + (size_t)(blockN + row) * K + k0 + kk);
      *(uint4*)(&Bs[row * LDSROW + kk]) = vb;
#endif
    }
#if USE_ASYNC_LDS
    asm volatile("s_wait_asynccnt 0x0" ::: "memory");
#endif
    __syncthreads();

    union Frag { v16bf v; uint4 q[2]; };
    #pragma unroll
    for (int ks = 0; ks < TKS; ks += 32) {
      Frag af[2], bfr[4];
      #pragma unroll
      for (int mf = 0; mf < 2; ++mf) {
        const uint16_t* p = &As[(waveM + mf * 16 + m) * LDSROW + ks + ka];
        af[mf].q[0] = *(const uint4*)(p);
        af[mf].q[1] = *(const uint4*)(p + 16);
      }
      #pragma unroll
      for (int nf = 0; nf < 4; ++nf) {
        const uint16_t* p = &Bs[(waveN + nf * 16 + m) * LDSROW + ks + kb];
        bfr[nf].q[0] = *(const uint4*)(p);
        bfr[nf].q[1] = *(const uint4*)(p + 16);
      }
      #pragma unroll
      for (int mf = 0; mf < 2; ++mf)
        #pragma unroll
        for (int nf = 0; nf < 4; ++nf)
          acc[mf][nf] = __builtin_amdgcn_wmma_f32_16x16x32_bf16(
              false, af[mf].v, false, bfr[nf].v, (short)0, acc[mf][nf], false, false);
    }
    __syncthreads();
  }

  // epilogue (C layout: lanes 0-15 -> rows 0-7, lanes 16-31 -> rows 8-15)
  char* outBase = (char*)Cout + (size_t)blockIdx.z * sliceBytes;
  #pragma unroll
  for (int mf = 0; mf < 2; ++mf) {
    const int rbase = blockM + waveM + mf * 16 + ((lane >> 4) * 8);
    #pragma unroll
    for (int nf = 0; nf < 4; ++nf) {
      const int col = blockN + waveN + nf * 16 + (lane & 15);
      const float bv = RAW ? 0.0f : bias[col];
      #pragma unroll
      for (int r = 0; r < 8; ++r) {
        float v = acc[mf][nf][r] + bv;
        if (!RAW) {
          if (ACT == 1)      v = v > 0.0f ? v : 0.0f;
          else if (ACT == 2) v = v / (1.0f + __expf(-v));
        }
        const size_t idx = (size_t)(rbase + r) * N + col;
        if (!RAW && OUTBF16) ((uint16_t*)outBase)[idx] = bf16_rn(v);
        else                 ((float*)outBase)[idx]    = v;
      }
    }
  }
}

// ---------------- split-K reduce: out = act(sum_s partial[s] + bias) ----------------
template <int ACT, int OUTBF16>
__global__ __launch_bounds__(256) void AH_splitk_reduce(const float* __restrict__ part,
                                                        size_t sliceElems, int S,
                                                        const float* __restrict__ bias,
                                                        void* __restrict__ out,
                                                        int N, int total) {
  const int i = blockIdx.x * 256 + threadIdx.x;
  if (i >= total) return;
  float v = 0.0f;
  for (int s = 0; s < S; ++s) v += part[(size_t)s * sliceElems + i];  // fixed order: deterministic
  v += bias[i % N];
  if (ACT == 1)      v = v > 0.0f ? v : 0.0f;
  else if (ACT == 2) v = v / (1.0f + __expf(-v));
  if (OUTBF16) ((uint16_t*)out)[i] = bf16_rn(v);
  else         ((float*)out)[i]    = v;
}

// ---------------- exact k-th-largest via MSB radix select (values >= 0) ----------------
__global__ __launch_bounds__(256) void AH_kwta(const float* __restrict__ X,
                                               uint16_t* __restrict__ Y,
                                               int Ncols, const int* __restrict__ kptr,
                                               int kconst) {
  extern __shared__ float srow[];          // Ncols floats
  __shared__ unsigned hist[256];
  __shared__ unsigned sh_prefix;
  __shared__ int sh_k;
  const int row = blockIdx.x;
  const int tid = threadIdx.x;
  const float* xr = X + (size_t)row * Ncols;
  for (int i = tid; i < Ncols; i += 256) srow[i] = xr[i];
  if (tid == 0) { sh_prefix = 0u; sh_k = kptr ? *kptr : kconst; }
  __syncthreads();
  for (int pass = 0; pass < 4; ++pass) {
    const int shift = 24 - pass * 8;
    hist[tid] = 0u;
    __syncthreads();
    const unsigned prefix = sh_prefix;
    const unsigned pmask  = (pass == 0) ? 0u : (0xffffffffu << (32 - pass * 8));
    for (int i = tid; i < Ncols; i += 256) {
      const unsigned u = __float_as_uint(srow[i]);
      if ((u & pmask) == prefix) atomicAdd(&hist[(u >> shift) & 255u], 1u);
    }
    __syncthreads();
    if (tid == 0) {
      int k = sh_k, cum = 0, bsel = 0;
      for (int b = 255; b >= 0; --b) {
        const int c = (int)hist[b];
        if (cum + c >= k) { bsel = b; break; }
        cum += c;
      }
      sh_prefix |= ((unsigned)bsel << shift);
      sh_k = k - cum;
    }
    __syncthreads();
  }
  const float thresh = __uint_as_float(sh_prefix);
  uint16_t* yr = Y + (size_t)row * Ncols;
  for (int i = tid; i < Ncols; i += 256) {
    const float v = srow[i];
    yr[i] = (v >= thresh) ? bf16_rn(v) : (uint16_t)0;
  }
}

// ---------------- per-row softmax entropy ----------------
__global__ __launch_bounds__(256) void AH_entropy(const float* __restrict__ H,
                                                  float* __restrict__ ent, int Ncols) {
  __shared__ float red[256];
  const int row = blockIdx.x, tid = threadIdx.x;
  const float* hr = H + (size_t)row * Ncols;
  float mx = -1e30f;
  for (int i = tid; i < Ncols; i += 256) mx = fmaxf(mx, hr[i]);
  red[tid] = mx; __syncthreads();
  for (int s = 128; s > 0; s >>= 1) { if (tid < s) red[tid] = fmaxf(red[tid], red[tid + s]); __syncthreads(); }
  mx = red[0]; __syncthreads();
  float se = 0.f, sxe = 0.f;
  for (int i = tid; i < Ncols; i += 256) {
    const float x = hr[i];
    const float e = __expf(x - mx);
    se += e; sxe += x * e;
  }
  red[tid] = se; __syncthreads();
  for (int s = 128; s > 0; s >>= 1) { if (tid < s) red[tid] += red[tid + s]; __syncthreads(); }
  se = red[0]; __syncthreads();
  red[tid] = sxe; __syncthreads();
  for (int s = 128; s > 0; s >>= 1) { if (tid < s) red[tid] += red[tid + s]; __syncthreads(); }
  sxe = red[0];
  if (tid == 0) ent[row] = (mx + __logf(se)) - sxe / se;
}

// ---------------- mean entropy -> k_dyn (deterministic fixed-order reduce) ----------------
__global__ __launch_bounds__(256) void AH_finalize_k(const float* __restrict__ ent, int rows,
                                                     int Hdim, int Kbase, int* __restrict__ kout) {
  __shared__ float red[256];
  const int tid = threadIdx.x;
  float s = 0.f;
  for (int i = tid; i < rows; i += 256) s += ent[i];
  red[tid] = s; __syncthreads();
  for (int st = 128; st > 0; st >>= 1) { if (tid < st) red[tid] += red[tid + st]; __syncthreads(); }
  if (tid == 0) {
    const float mean = red[0] / (float)rows;
    int k = (int)((float)Kbase * (1.0f + 0.5f * mean / __logf((float)Hdim)));
    if (k > Hdim / 2) k = Hdim / 2;
    *kout = k;
  }
}

// ---------------- host launch ----------------
extern "C" void kernel_launch(void* const* d_in, const int* in_sizes, int n_in,
                              void* d_out, int out_size, void* d_ws, size_t ws_size,
                              hipStream_t stream) {
  const int B = 2048, Q = 2048, E = 1024, Z = 10240, D = 512, H = 4096;

  const float* query   = (const float*)d_in[0];
  const float* W_ent   = (const float*)d_in[1];
  const float* b_ent   = (const float*)d_in[2];
  const float* W_exp   = (const float*)d_in[3];
  const float* b_exp   = (const float*)d_in[4];
  const float* W_store = (const float*)d_in[5];
  const float* b_store = (const float*)d_in[6];
  const float* W_enc   = (const float*)d_in[7];
  const float* b_enc   = (const float*)d_in[8];
  const float* W_dec   = (const float*)d_in[9];
  const float* b_dec   = (const float*)d_in[10];

  char* p = (char*)d_ws;
  auto take = [&](size_t bytes) { char* r = p; p += (bytes + 255) & ~(size_t)255; return r; };
  uint16_t* q_bf     = (uint16_t*)take((size_t)B * Q * 2);
  uint16_t* Wt_ent   = (uint16_t*)take((size_t)Q * E * 2);
  uint16_t* Wt_exp   = (uint16_t*)take((size_t)E * Z * 2);
  uint16_t* Wt_store = (uint16_t*)take((size_t)Z * D * 2);
  uint16_t* Wt_enc   = (uint16_t*)take((size_t)D * H * 2);
  uint16_t* Wt_dec   = (uint16_t*)take((size_t)H * D * 2);
  uint16_t* ent_bf   = (uint16_t*)take((size_t)B * E * 2);
  float*    z_f32    = (float*)   take((size_t)B * Z * 4);
  uint16_t* zs_bf    = (uint16_t*)take((size_t)B * Z * 2);
  uint16_t* x_bf     = (uint16_t*)take((size_t)B * D * 2);
  float*    h_f32    = (float*)   take((size_t)B * H * 4);
  uint16_t* hs_bf    = (uint16_t*)take((size_t)B * H * 2);
  float*    entarr   = (float*)   take((size_t)B * 4);
  int*      kdyn     = (int*)     take(256);
  float*    partial  = (float*)   take((size_t)8 * B * D * 4);  // max split-K partials (33.5MB)

  // preprocessing: query -> bf16, weights -> transposed bf16
  AH_cvt_bf16<<<(B * Q + 255) / 256, 256, 0, stream>>>(query, q_bf, B * Q);
  AH_transpose_bf16<<<dim3(E / 32, Q / 32), 256, 0, stream>>>(W_ent,   Wt_ent,   Q, E);
  AH_transpose_bf16<<<dim3(Z / 32, E / 32), 256, 0, stream>>>(W_exp,   Wt_exp,   E, Z);
  AH_transpose_bf16<<<dim3(D / 32, Z / 32), 256, 0, stream>>>(W_store, Wt_store, Z, D);
  AH_transpose_bf16<<<dim3(H / 32, D / 32), 256, 0, stream>>>(W_enc,   Wt_enc,   D, H);
  AH_transpose_bf16<<<dim3(D / 32, H / 32), 256, 0, stream>>>(W_dec,   Wt_dec,   H, D);

  // ent = silu(query @ W_ent + b_ent)  [B,E]: split-K S=2 (256 WGs) -> reduce -> bf16
  AH_gemm_bf16<0, 0, 1><<<dim3(E / TN, B / TM, 2), 256, 0, stream>>>(
      q_bf, Wt_ent, nullptr, partial, B, E, Q, Q / 2, (size_t)B * E * 4);
  AH_splitk_reduce<2, 1><<<(B * E + 255) / 256, 256, 0, stream>>>(
      partial, (size_t)B * E, 2, b_ent, ent_bf, E, B * E);

  // z = relu(ent @ W_exp + b_exp)  [B,Z]: direct (1280 WGs)
  AH_gemm_bf16<1, 0, 0><<<dim3(Z / TN, B / TM, 1), 256, 0, stream>>>(
      ent_bf, Wt_exp, b_exp, z_f32, B, Z, E, E, 0);

  // static kWTA (k=512) -> bf16
  AH_kwta<<<B, 256, (size_t)Z * 4, stream>>>(z_f32, zs_bf, Z, nullptr, 512);

  // x = silu(z_sparse @ W_store + b_store)  [B,D]: split-K S=8 (512 WGs)
  AH_gemm_bf16<0, 0, 1><<<dim3(D / TN, B / TM, 8), 256, 0, stream>>>(
      zs_bf, Wt_store, nullptr, partial, B, D, Z, Z / 8, (size_t)B * D * 4);
  AH_splitk_reduce<2, 1><<<(B * D + 255) / 256, 256, 0, stream>>>(
      partial, (size_t)B * D, 8, b_store, x_bf, D, B * D);

  // h = relu(x @ W_enc + b_enc)  [B,H]: direct (512 WGs)
  AH_gemm_bf16<1, 0, 0><<<dim3(H / TN, B / TM, 1), 256, 0, stream>>>(
      x_bf, Wt_enc, b_enc, h_f32, B, H, D, D, 0);

  // entropy-adaptive k
  AH_entropy<<<B, 256, 0, stream>>>(h_f32, entarr, H);
  AH_finalize_k<<<1, 256, 0, stream>>>(entarr, B, H, 512, kdyn);

  // dynamic kWTA -> bf16
  AH_kwta<<<B, 256, (size_t)H * 4, stream>>>(h_f32, hs_bf, H, kdyn, 0);

  // recon = h_sparse @ W_dec + b_dec  [B,D]: split-K S=4 (256 WGs) -> f32 d_out
  AH_gemm_bf16<0, 0, 1><<<dim3(D / TN, B / TM, 4), 256, 0, stream>>>(
      hs_bf, Wt_dec, nullptr, partial, B, D, H, H / 4, (size_t)B * D * 4);
  AH_splitk_reduce<0, 0><<<(B * D + 255) / 256, 256, 0, stream>>>(
      partial, (size_t)B * D, 4, b_dec, (float*)d_out, D, B * D);

  (void)in_sizes; (void)n_in; (void)out_size; (void)ws_size;
}
